// GAT_41832981463436
// MI455X (gfx1250) — matrix operand order repaired and checked by reading, not compiled, not run on hardware
//
#include <hip/hip_runtime.h>
#include <hip/hip_bf16.h>

typedef _Float16 v16h __attribute__((ext_vector_type(16)));
typedef _Float16 v8h  __attribute__((ext_vector_type(8)));
typedef float    v8f  __attribute__((ext_vector_type(8)));

#define N_NODES 50000
#define FEAT    128

// ---------------- degree / normalization ----------------
__global__ void k_zero(float* __restrict__ p, int n) {
    int i = blockIdx.x * blockDim.x + threadIdx.x;
    if (i < n) p[i] = 0.0f;
}

__global__ void k_deg(const int* __restrict__ dst, float* __restrict__ deg, int E) {
    int e = blockIdx.x * blockDim.x + threadIdx.x;
    if (e < E) atomicAdd(&deg[dst[e]], 1.0f);
}

__global__ void k_inv(float* __restrict__ d, int N) {
    int i = blockIdx.x * blockDim.x + threadIdx.x;
    if (i < N) d[i] = rsqrtf(d[i] + 1.0f);   // +1 self-loop; deg>0 always
}

// ---------------- f32 -> f16 convert with K padding ----------------
__global__ void k_cvt_pad(const float* __restrict__ in, _Float16* __restrict__ out,
                          int rows, int K, int Kpad) {
    int i = blockIdx.x * blockDim.x + threadIdx.x;
    if (i >= rows * Kpad) return;
    int r = i / Kpad;
    int k = i - r * Kpad;
    float v = (k < K) ? in[(size_t)r * K + k] : 0.0f;
    out[i] = (_Float16)v;
}

// ---------------- WMMA GEMM: C[M,Nout] = A[M,Kpad](f16) x B[Nout,Kpad](f16)^T ----
// One wave per 16x16 tile; block = 256 threads = 8 waves = one M-tile x Nout=128.
__global__ void k_wmma_gemm(const _Float16* __restrict__ A,
                            const _Float16* __restrict__ B,
                            float* __restrict__ C,
                            int M, int Kpad, int Nout) {
    const int lane = threadIdx.x & 31;
    const int wave = threadIdx.x >> 5;        // n-tile 0..7
    const int m0   = blockIdx.x * 16;
    const int n0   = wave * 16;
    if (m0 >= M || n0 >= Nout) return;        // wave-uniform; EXEC stays all-1s
    const int row = lane & 15;
    const int hi  = lane >> 4;                // half-wave select

    // A frag (16x32, MxK): lanes0-15 K = {k0..k0+7, k0+16..k0+23}; lanes16-31 +8
    const _Float16* arow = A + (size_t)(m0 + row) * Kpad + hi * 8;
    // B frag (32x16, KxN): same K order as A: lanes0-15 K = {k0..k0+7, k0+16..k0+23}; lanes16-31 +8
    const _Float16* brow = B + (size_t)(n0 + row) * Kpad + hi * 8;

    v8f acc = {0.f, 0.f, 0.f, 0.f, 0.f, 0.f, 0.f, 0.f};
    for (int k0 = 0; k0 < Kpad; k0 += 32) {
        union { v16h v; v8h h[2]; } a, b;
        a.h[0] = *(const v8h*)(arow + k0);        // 16B-aligned global_load_b128
        a.h[1] = *(const v8h*)(arow + k0 + 16);
        b.h[0] = *(const v8h*)(brow + k0);
        b.h[1] = *(const v8h*)(brow + k0 + 16);
        acc = __builtin_amdgcn_wmma_f32_16x16x32_f16(
                  false, a.v, false, b.v, (short)0, acc, false, false);
    }

    // D layout: N = lane&15 ; M = hi*8 + r for VGPR r
    float* crow = C + (size_t)(m0 + hi * 8) * Nout + n0 + row;
#pragma unroll
    for (int r = 0; r < 8; ++r)
        crow[(size_t)r * Nout] = acc[r];
}

// ---------------- aggregation: self-loop init then edge scatter-add ------------
__global__ void k_self(const float* __restrict__ h, const float* __restrict__ inv,
                       float* __restrict__ out, int N) {
    int i = blockIdx.x * blockDim.x + threadIdx.x;
    if (i >= N * FEAT) return;
    int r = i >> 7;
    float w = inv[r];
    out[i] = w * w * h[i];
}

__global__ void k_scatter(const int* __restrict__ src, const int* __restrict__ dst,
                          const float* __restrict__ inv, const float* __restrict__ h,
                          float* __restrict__ out, int E) {
    int t = blockIdx.x * blockDim.x + threadIdx.x;   // E*32 = 51.2M < 2^31
    if (t >= E * 32) return;
    int e = t >> 5;            // one edge per 32 threads (coalesced 128 floats)
    int g = t & 31;
    int s = src[e];
    int d = dst[e];
    float nrm = inv[s] * inv[d];
    const float4 v = *(const float4*)(h + ((size_t)s << 7) + (g << 2));
    float* o = out + ((size_t)d << 7) + (g << 2);
    atomicAdd(o + 0, v.x * nrm);
    atomicAdd(o + 1, v.y * nrm);
    atomicAdd(o + 2, v.z * nrm);
    atomicAdd(o + 3, v.w * nrm);
}

__global__ void k_relu_h(const float* __restrict__ in, _Float16* __restrict__ out, int n) {
    int i = blockIdx.x * blockDim.x + threadIdx.x;
    if (i < n) out[i] = (_Float16)fmaxf(in[i], 0.0f);
}

// ---------------- final linear [N,128] x [16,128]^T -> [N,16] (f32 VALU) -------
__global__ void k_lin(const float* __restrict__ h, const float* __restrict__ W,
                      float* __restrict__ out, int N) {
    int t = blockIdx.x * blockDim.x + threadIdx.x;
    if (t >= N * 16) return;
    int i = t >> 4;
    int c = t & 15;
    const float* hr = h + ((size_t)i << 7);
    const float* wr = W + (c << 7);
    float acc = 0.0f;
#pragma unroll
    for (int k = 0; k < 128; ++k) acc += hr[k] * wr[k];
    out[t] = acc;
}

extern "C" void kernel_launch(void* const* d_in, const int* in_sizes, int n_in,
                              void* d_out, int out_size, void* d_ws, size_t ws_size,
                              hipStream_t stream) {
    const float* x    = (const float*)d_in[0];
    const int*   ei   = (const int*)d_in[1];
    const float* W1   = (const float*)d_in[2];
    const float* W2   = (const float*)d_in[3];
    const float* Wlin = (const float*)d_in[4];
    float* out = (float*)d_out;

    const int N  = N_NODES;
    const int E  = in_sizes[1] / 2;
    const int K1 = 500, K1P = 512;
    const int* src  = ei;
    const int* dstE = ei + E;

    // workspace carve-out (256B aligned regions)
    char* ws = (char*)d_ws;
    size_t off = 0;
    auto carve = [&](size_t bytes) -> void* {
        void* p = ws + off;
        off += (bytes + 255) & ~(size_t)255;
        return p;
    };
    float*    inv  = (float*)   carve((size_t)N * 4);
    _Float16* xh   = (_Float16*)carve((size_t)N * K1P * 2);
    _Float16* w1h  = (_Float16*)carve((size_t)FEAT * K1P * 2);
    _Float16* w2h  = (_Float16*)carve((size_t)FEAT * FEAT * 2);
    float*    hpre = (float*)   carve((size_t)N * FEAT * 4);
    float*    hagg = (float*)   carve((size_t)N * FEAT * 4);
    _Float16* hh   = (_Float16*)carve((size_t)N * FEAT * 2);

    const int T = 256;
    auto g = [](long long n, int t) { return (unsigned)((n + t - 1) / t); };

    // normalization: deg on dst (+1 self loop) -> rsqrt
    k_zero<<<g(N, T), T, 0, stream>>>(inv, N);
    k_deg<<<g(E, T), T, 0, stream>>>(dstE, inv, E);
    k_inv<<<g(N, T), T, 0, stream>>>(inv, N);

    // f16 conversions (pad K 500->512 with zeros)
    k_cvt_pad<<<g((long long)N * K1P, T), T, 0, stream>>>(x, xh, N, K1, K1P);
    k_cvt_pad<<<g(FEAT * K1P, T), T, 0, stream>>>(W1, w1h, FEAT, K1, K1P);
    k_cvt_pad<<<g(FEAT * FEAT, T), T, 0, stream>>>(W2, w2h, FEAT, FEAT, FEAT);

    // layer 1: h = x @ W1^T  (WMMA), then normalized aggregation
    k_wmma_gemm<<<N / 16, T, 0, stream>>>(xh, w1h, hpre, N, K1P, FEAT);
    k_self<<<g((long long)N * FEAT, T), T, 0, stream>>>(hpre, inv, hagg, N);
    k_scatter<<<g((long long)E * 32, T), T, 0, stream>>>(src, dstE, inv, hpre, hagg, E);
    k_relu_h<<<g((long long)N * FEAT, T), T, 0, stream>>>(hagg, hh, N * FEAT);

    // layer 2: h = relu1 @ W2^T (WMMA), then normalized aggregation
    k_wmma_gemm<<<N / 16, T, 0, stream>>>(hh, w2h, hpre, N, FEAT, FEAT);
    k_self<<<g((long long)N * FEAT, T), T, 0, stream>>>(hpre, inv, hagg, N);
    k_scatter<<<g((long long)E * 32, T), T, 0, stream>>>(src, dstE, inv, hpre, hagg, E);

    // final classifier (f32 for output precision)
    k_lin<<<g((long long)N * 16, T), T, 0, stream>>>(hagg, Wlin, out, N);
}
